// SEAL_31198642438218
// MI455X (gfx1250) — matrix operand order repaired and checked by reading, not compiled, not run on hardware
//
#include <hip/hip_runtime.h>
#include <hip/hip_bf16.h>

typedef __attribute__((ext_vector_type(16))) _Float16 v16h;
typedef __attribute__((ext_vector_type(8)))  _Float16 v8h;
typedef __attribute__((ext_vector_type(8)))  float    v8f;

#define IN_DIM 18
#define HID    64
#define EMB    32

static __device__ __forceinline__ v16h cat8(v8h lo, v8h hi) {
    return __builtin_shufflevector(lo, hi, 0,1,2,3,4,5,6,7,8,9,10,11,12,13,14,15);
}

// ---------------------------------------------------------------------------
// degree / norm
// ---------------------------------------------------------------------------
__global__ void k_deg_init(float* deg, int n) {
    int i = blockIdx.x * blockDim.x + threadIdx.x;
    if (i < n) deg[i] = 1.0f;               // self-loop contributes 1
}

__global__ void k_deg_edges(const int* __restrict__ ei, float* deg, int nE) {
    int e = blockIdx.x * blockDim.x + threadIdx.x;
    if (e < nE) atomicAdd(&deg[ei[nE + e]], 1.0f);   // dst = ei[1][e]
}

__global__ void k_deg_to_dis(float* deg, int n) {
    int i = blockIdx.x * blockDim.x + threadIdx.x;
    if (i < n) deg[i] = rsqrtf(deg[i]);     // in-place: deg -> D^{-1/2}
}

// ---------------------------------------------------------------------------
// operand prep: zero-padded f16 copies (K-contiguous) so GEMM loads are b128
// ---------------------------------------------------------------------------
__global__ void k_prep_x(const float* __restrict__ x, _Float16* __restrict__ xpad,
                         int n, int npad) {
    int t = blockIdx.x * blockDim.x + threadIdx.x;
    if (t >= npad * 32) return;
    int i = t >> 5, k = t & 31;
    xpad[t] = (i < n && k < IN_DIM) ? (_Float16)x[(size_t)i * IN_DIM + k]
                                    : (_Float16)0.f;
}

__global__ void k_prep_w1(const float* __restrict__ W1, _Float16* __restrict__ w1t) {
    int t = blockIdx.x * blockDim.x + threadIdx.x;   // [64 cols][32 K]
    if (t >= HID * 32) return;
    int col = t >> 5, k = t & 31;
    w1t[t] = (k < IN_DIM) ? (_Float16)W1[(size_t)k * HID + col] : (_Float16)0.f;
}

__global__ void k_prep_w2(const float* __restrict__ W2, _Float16* __restrict__ w2t) {
    int t = blockIdx.x * blockDim.x + threadIdx.x;   // [32 cols][64 K]
    if (t >= EMB * HID) return;
    int col = t >> 6, k = t & 63;
    w2t[t] = (_Float16)W2[(size_t)k * EMB + col];
}

// ---------------------------------------------------------------------------
// GEMM1: hw1[npad,64] = xpad[npad,32] @ W1pad[32,64]; one wave32 per 16-row
// tile. Output buffer has npad rows -> stores are unguarded (pad rows are
// never read downstream).
// ---------------------------------------------------------------------------
__global__ void k_gemm1_wmma(const _Float16* __restrict__ xpad,
                             const _Float16* __restrict__ w1t,
                             float* __restrict__ hw1, int npad) {
    int wave = (blockIdx.x * blockDim.x + threadIdx.x) >> 5;
    int lane = threadIdx.x & 31;
    int row0 = wave * 16;
    if (row0 >= npad) return;               // wave-uniform
    int half = lane >> 4;
    int m    = lane & 15;

    // A 16x32 f16: lane(half,m) holds K {half*16..+15} contiguous
    const _Float16* xr = xpad + (size_t)(row0 + m) * 32;
    v16h a = cat8(*(const v8h*)(xr + half * 16),
                  *(const v8h*)(xr + half * 16 + 8));

#pragma unroll
    for (int c = 0; c < 4; ++c) {           // HID = 64 -> 4 N-chunks of 16
        int ncol = c * 16 + m;
        // B 32x16 f16: lane half selects K 0-15 / 16-31, contiguous in w1t
        const _Float16* wc = w1t + (size_t)ncol * 32 + half * 16;
        v16h b = cat8(*(const v8h*)(wc), *(const v8h*)(wc + 8));
        v8f acc = {};
        acc = __builtin_amdgcn_wmma_f32_16x16x32_f16(false, a, false, b,
                                                     (short)0, acc, false, false);
        // C/D: VGPR r -> M = r + 8*half, N = m ; immediate-offset row stores
        float* outp = hw1 + (size_t)(row0 + 8 * half) * HID + ncol;
#pragma unroll
        for (int r = 0; r < 8; ++r)
            outp[r * HID] = acc[r];
    }
}

// ---------------------------------------------------------------------------
// aggregation layer 1
// ---------------------------------------------------------------------------
__global__ void k_selfloop1(const float* __restrict__ hw1,
                            const float* __restrict__ dis,
                            float* __restrict__ agg1, int n) {
    int t = blockIdx.x * blockDim.x + threadIdx.x;
    if (t >= n * HID) return;
    int i = t >> 6;
    float s = dis[i];
    agg1[t] = hw1[t] * s * s;               // initializes agg1 (no memset needed)
}

__global__ void k_scatter1(const int* __restrict__ ei,
                           const float* __restrict__ dis,
                           const float* __restrict__ hw1,
                           float* __restrict__ agg1, int nE) {
    long long t = (long long)blockIdx.x * blockDim.x + threadIdx.x;
    long long e = t >> 5;                   // one wave32 per edge, 2 floats/lane
    int lane = (int)(t & 31);
    if (e >= nE) return;
    int s = ei[e], d = ei[(long long)nE + e];
    float nrm = dis[s] * dis[d];
    float2 v = ((const float2*)(hw1 + (size_t)s * HID))[lane];
    float* dstp = agg1 + (size_t)d * HID + lane * 2;
    atomicAdd(dstp,     v.x * nrm);
    atomicAdd(dstp + 1, v.y * nrm);
}

// relu+bias, emit f16 rows (zero pad rows >= n) for the WMMA consumer
__global__ void k_relu_bias(const float* __restrict__ agg1,
                            const float* __restrict__ b1,
                            _Float16* __restrict__ hpad, int n, int npad) {
    int t = blockIdx.x * blockDim.x + threadIdx.x;
    if (t >= npad * HID) return;
    int i = t >> 6, f = t & 63;
    float v = 0.f;
    if (i < n) {
        v = agg1[t] + b1[f];
        v = v > 0.f ? v : 0.f;
    }
    hpad[t] = (_Float16)v;
}

// ---------------------------------------------------------------------------
// GEMM2: hw2[npad,32] = hpad[npad,64] @ W2[64,32]; 2 K-steps x 2 N-chunks
// ---------------------------------------------------------------------------
__global__ void k_gemm2_wmma(const _Float16* __restrict__ hpad,
                             const _Float16* __restrict__ w2t,
                             float* __restrict__ hw2, int npad) {
    int wave = (blockIdx.x * blockDim.x + threadIdx.x) >> 5;
    int lane = threadIdx.x & 31;
    int row0 = wave * 16;
    if (row0 >= npad) return;
    int half = lane >> 4;
    int m    = lane & 15;
    const _Float16* hr = hpad + (size_t)(row0 + m) * HID;

    v8f acc[2] = {{}, {}};
#pragma unroll
    for (int ks = 0; ks < HID; ks += 32) {  // K = 64 -> 2 WMMA K-steps
        v16h a = cat8(*(const v8h*)(hr + ks + half * 16),
                      *(const v8h*)(hr + ks + half * 16 + 8));
#pragma unroll
        for (int c = 0; c < 2; ++c) {       // EMB = 32 -> 2 N-chunks
            const _Float16* wc = w2t + (size_t)(c * 16 + m) * HID + ks + half * 16;
            v16h b = cat8(*(const v8h*)(wc), *(const v8h*)(wc + 8));
            acc[c] = __builtin_amdgcn_wmma_f32_16x16x32_f16(false, a, false, b,
                                                            (short)0, acc[c],
                                                            false, false);
        }
    }
#pragma unroll
    for (int c = 0; c < 2; ++c) {
        float* outp = hw2 + (size_t)(row0 + 8 * half) * EMB + c * 16 + m;
#pragma unroll
        for (int r = 0; r < 8; ++r)
            outp[r * EMB] = acc[c][r];
    }
}

// ---------------------------------------------------------------------------
// aggregation layer 2 (EMB = 32 -> one float per lane)
// ---------------------------------------------------------------------------
__global__ void k_selfloop2(const float* __restrict__ hw2,
                            const float* __restrict__ dis,
                            float* __restrict__ agg2, int n) {
    int t = blockIdx.x * blockDim.x + threadIdx.x;
    if (t >= n * EMB) return;
    int i = t >> 5;
    float s = dis[i];
    agg2[t] = hw2[t] * s * s;
}

__global__ void k_scatter2(const int* __restrict__ ei,
                           const float* __restrict__ dis,
                           const float* __restrict__ hw2,
                           float* __restrict__ agg2, int nE) {
    long long t = (long long)blockIdx.x * blockDim.x + threadIdx.x;
    long long e = t >> 5;
    int lane = (int)(t & 31);
    if (e >= nE) return;
    int s = ei[e], d = ei[(long long)nE + e];
    float nrm = dis[s] * dis[d];
    float v = hw2[(size_t)s * EMB + lane];
    atomicAdd(&agg2[(size_t)d * EMB + lane], v * nrm);
}

// ---------------------------------------------------------------------------
// head: out[b] = sum_f (z[t0][f] * z[t1][f] * Wo[f]) + bo ; z = agg2 + b2
// ---------------------------------------------------------------------------
__global__ void k_head(const int* __restrict__ tg,
                       const float* __restrict__ agg2,
                       const float* __restrict__ b2,
                       const float* __restrict__ Wo,
                       const float* __restrict__ bo,
                       float* __restrict__ out, int batch) {
    int w = (blockIdx.x * blockDim.x + threadIdx.x) >> 5;
    int lane = threadIdx.x & 31;
    if (w >= batch) return;
    int t0 = tg[w], t1 = tg[batch + w];
    float bb = b2[lane];
    float z0 = agg2[(size_t)t0 * EMB + lane] + bb;
    float z1 = agg2[(size_t)t1 * EMB + lane] + bb;
    float p = z0 * z1 * Wo[lane];
#pragma unroll
    for (int off = 16; off > 0; off >>= 1)
        p += __shfl_down(p, off, 32);
    if (lane == 0) out[w] = p + bo[0];
}

// ---------------------------------------------------------------------------
extern "C" void kernel_launch(void* const* d_in, const int* in_sizes, int n_in,
                              void* d_out, int out_size, void* d_ws, size_t ws_size,
                              hipStream_t stream) {
    const float* x  = (const float*)d_in[0];
    const int*   ei = (const int*)  d_in[1];
    const int*   tg = (const int*)  d_in[2];
    const float* W1 = (const float*)d_in[3];
    const float* b1 = (const float*)d_in[4];
    const float* W2 = (const float*)d_in[5];
    const float* b2 = (const float*)d_in[6];
    const float* Wo = (const float*)d_in[7];
    const float* bo = (const float*)d_in[8];
    float* out = (float*)d_out;

    const int n     = in_sizes[0] / IN_DIM;
    const int nE    = in_sizes[1] / 2;
    const int batch = in_sizes[2] / 2;
    const int npad  = (n + 15) & ~15;

    // workspace (16B-aligned sections):
    // dis[nA] | hw1[npad*64] | agg1[npad*64] | hw2[npad*32] | agg2[npad*32] |
    // xpad f16[npad*32] | hpad f16[npad*64] | w1t f16[2048] | w2t f16[2048]
    const size_t nA = (size_t)((n + 3) & ~3);
    float* dis  = (float*)d_ws;
    float* hw1  = dis  + nA;
    float* agg1 = hw1  + (size_t)npad * HID;
    float* hw2  = agg1 + (size_t)npad * HID;
    float* agg2 = hw2  + (size_t)npad * EMB;
    _Float16* xpad = (_Float16*)(agg2 + (size_t)npad * EMB);
    _Float16* hpad = xpad + (size_t)npad * 32;
    _Float16* w1t  = hpad + (size_t)npad * HID;
    _Float16* w2t  = w1t + HID * 32;

    const int B = 256;
    const int tiles = npad / 16;

    k_deg_init  <<<(n + B - 1) / B, B, 0, stream>>>(dis, n);
    k_deg_edges <<<(nE + B - 1) / B, B, 0, stream>>>(ei, dis, nE);
    k_deg_to_dis<<<(n + B - 1) / B, B, 0, stream>>>(dis, n);

    k_prep_x <<<(npad * 32 + B - 1) / B, B, 0, stream>>>(x, xpad, n, npad);
    k_prep_w1<<<(HID * 32 + B - 1) / B, B, 0, stream>>>(W1, w1t);
    k_prep_w2<<<(EMB * HID + B - 1) / B, B, 0, stream>>>(W2, w2t);

    k_gemm1_wmma<<<(tiles * 32 + B - 1) / B, B, 0, stream>>>(xpad, w1t, hw1, npad);

    k_selfloop1 <<<(n * HID + B - 1) / B, B, 0, stream>>>(hw1, dis, agg1, n);
    {
        long long th = (long long)nE * 32;
        k_scatter1<<<(unsigned)((th + B - 1) / B), B, 0, stream>>>(ei, dis, hw1, agg1, nE);
    }
    k_relu_bias <<<(npad * HID + B - 1) / B, B, 0, stream>>>(agg1, b1, hpad, n, npad);

    k_gemm2_wmma<<<(tiles * 32 + B - 1) / B, B, 0, stream>>>(hpad, w2t, hw2, npad);

    k_selfloop2 <<<(n * EMB + B - 1) / B, B, 0, stream>>>(hw2, dis, agg2, n);
    {
        long long th = (long long)nE * 32;
        k_scatter2<<<(unsigned)((th + B - 1) / B), B, 0, stream>>>(ei, dis, hw2, agg2, nE);
    }

    k_head<<<(batch * 32 + B - 1) / B, B, 0, stream>>>(tg, agg2, b2, Wo, bo, out, batch);
}
